// MessagePassingModel_10385230922634
// MI455X (gfx1250) — matrix operand order repaired and checked by reading, not compiled, not run on hardware
//
#include <hip/hip_runtime.h>

typedef __bf16 bf16_t;
typedef __attribute__((ext_vector_type(16))) __bf16 v16bf;
typedef __attribute__((ext_vector_type(8)))  float  v8f;

#define NF 64
#define NK 32
#define NL 9
#define NA 21
#define NMOL 512
#define NATOM (NMOL * NA)
#define EPM (NA * (NA - 1))          /* 420 edges per molecule */
#define ETOT ((long)NMOL * EPM)
#define ETILES ((EPM + 15) / 16)     /* 27 */
#define NWAVE 8
#define CUTR 3.0f

// spherical channel l -> degree group (compile-time foldable)
__host__ __device__ constexpr int degc(int l) { return l == 0 ? 0 : (l < 4 ? 1 : 2); }
// degree group -> contiguous l range {0},{1,2,3},{4..8}
__host__ __device__ constexpr int ch_lo(int c) { return c == 0 ? 0 : (c == 1 ? 1 : 4); }
__host__ __device__ constexpr int ch_hi(int c) { return c == 0 ? 1 : (c == 1 ? 4 : 9); }

// ---------------------------------------------------------------------------
// WMMA fragment helpers (layouts per CDNA5 ISA 7.12.2, wave32).
// ---------------------------------------------------------------------------
union frag16 {
  uint4 q[2];
  v16bf v;
};
__device__ __forceinline__ v8f vzero8() {
  v8f z = {0.f, 0.f, 0.f, 0.f, 0.f, 0.f, 0.f, 0.f};
  return z;
}
__device__ __forceinline__ v8f wmma_bf(v16bf a, v16bf b, v8f c) {
  return __builtin_amdgcn_wmma_f32_16x16x32_bf16(false, a, false, b, (short)0, c,
                                                 false, false);
}
// A fragment (16x32 bf16, row-major p[row*ld + koff + k]); rows >= rowLimit -> 0.
__device__ __forceinline__ v16bf load_a_bf(const bf16_t* p, int ld, int koff,
                                           int rowLimit) {
  int lane = threadIdx.x & 31;
  int row = lane & 15;
  int kb = (lane >> 4) << 3;
  const bf16_t* r0 = p + row * ld + koff + kb;
  frag16 f;
  if (row < rowLimit) {
    f.q[0] = *(const uint4*)(r0);
    f.q[1] = *(const uint4*)(r0 + 16);
  } else {
    f.q[0] = uint4{0u, 0u, 0u, 0u};
    f.q[1] = uint4{0u, 0u, 0u, 0u};
  }
  return f.v;
}
// A fragment from f32 channel tensor: row r=a*LCH+l -> p[(a*NL+l)*NF + koff + k]
template <int LCH>
__device__ __forceinline__ v16bf load_a_f32ch(const float* p, int r0, int koff,
                                              int rows) {
  int lane = threadIdx.x & 31;
  int r = r0 + (lane & 15);
  int kb = (lane >> 4) << 3;
  if (r < rows) {
    int ai = r / LCH, l = r - ai * LCH;
    const float* b0 = p + (ai * NL + l) * NF + koff + kb;
    float4 f0 = *(const float4*)(b0);
    float4 f1 = *(const float4*)(b0 + 4);
    float4 f2 = *(const float4*)(b0 + 16);
    float4 f3 = *(const float4*)(b0 + 20);
    v16bf a;
    a[0] = (bf16_t)f0.x;  a[1] = (bf16_t)f0.y;  a[2] = (bf16_t)f0.z;  a[3] = (bf16_t)f0.w;
    a[4] = (bf16_t)f1.x;  a[5] = (bf16_t)f1.y;  a[6] = (bf16_t)f1.z;  a[7] = (bf16_t)f1.w;
    a[8] = (bf16_t)f2.x;  a[9] = (bf16_t)f2.y;  a[10] = (bf16_t)f2.z; a[11] = (bf16_t)f2.w;
    a[12] = (bf16_t)f3.x; a[13] = (bf16_t)f3.y; a[14] = (bf16_t)f3.z; a[15] = (bf16_t)f3.w;
    return a;
  }
  frag16 f;
  f.q[0] = uint4{0u, 0u, 0u, 0u};
  f.q[1] = uint4{0u, 0u, 0u, 0u};
  return f.v;
}
// B fragment: column n = lane&15, K = kb..kb+7 and kb+16..kb+23 (kb = 8*(lane/16))
__device__ __forceinline__ v16bf load_b_bf(const bf16_t* p, int ncol0, int ld,
                                           int koff) {
  int lane = threadIdx.x & 31;
  const bf16_t* r0 = p + (ncol0 + (lane & 15)) * ld + koff + ((lane >> 4) << 3);
  frag16 f;
  f.q[0] = *(const uint4*)(r0);
  f.q[1] = *(const uint4*)(r0 + 16);
  return f.v;
}

// Per-lane edge->atom LDS offsets for this lane's 8 C-rows, kept in registers.
__device__ __forceinline__ void load_edge_offs(const int* srcg, const int* dstg,
                                               long gb, int e0, int mhalf,
                                               int cnt, int abase, int* soff,
                                               int* doff) {
  if (cnt == 16) {
    int4 sa = *(const int4*)(srcg + gb + e0 + mhalf);
    int4 sb = *(const int4*)(srcg + gb + e0 + mhalf + 4);
    int4 da = *(const int4*)(dstg + gb + e0 + mhalf);
    int4 db = *(const int4*)(dstg + gb + e0 + mhalf + 4);
    int s[8] = {sa.x, sa.y, sa.z, sa.w, sb.x, sb.y, sb.z, sb.w};
    int d[8] = {da.x, da.y, da.z, da.w, db.x, db.y, db.z, db.w};
#pragma unroll
    for (int v = 0; v < 8; ++v) {
      soff[v] = (s[v] - abase) * (NL * NF);
      doff[v] = (d[v] - abase) * (NL * NF);
    }
  } else {
#pragma unroll
    for (int v = 0; v < 8; ++v) {
      int m = mhalf + v;
      long e = gb + e0 + m;
      int s = (m < cnt) ? srcg[e] : abase;
      int d = (m < cnt) ? dstg[e] : abase;
      soff[v] = (s - abase) * (NL * NF);
      doff[v] = (d - abase) * (NL * NF);
    }
  }
}

// ---------------------------------------------------------------------------
// Edge message forward: ys[dst,l,f] += rexp*Y*s0 + rself*x_src (fused add).
// Padded rows: zeroed A rows + Y=0 + dummy atom 0 -> exact zero adds.
// ---------------------------------------------------------------------------
template <int NCH, int LCH>
__device__ void edge_forward(int mol, const bf16_t* Rbf, const float* Yf,
                             const int* srcg, const int* dstg, const bf16_t* wT,
                             const float* xs, float* ys, float* Yt) {
  int wv = threadIdx.x >> 5, lane = threadIdx.x & 31;
  int mhalf = (lane >> 4) << 3;
  long gb = (long)mol * EPM;
  int abase = mol * NA;
  float* Ytw = Yt + wv * 16 * NL;
  for (int tile = wv; tile < ETILES; tile += NWAVE) {
    int e0 = tile * 16;
    int cnt = (EPM - e0 < 16) ? (EPM - e0) : 16;
    if (tile + NWAVE < ETILES)
      __builtin_prefetch(Rbf + (gb + e0 + 16 * NWAVE) * NK, 0, 0);
    if (lane < 16) {
      int m = lane;
      if (m < cnt) {
        long e = gb + e0 + m;
#pragma unroll
        for (int l = 0; l < NL; ++l) Ytw[m * NL + l] = Yf[e * NL + l];
      } else {
#pragma unroll
        for (int l = 0; l < NL; ++l) Ytw[m * NL + l] = 0.f;
      }
    }
    int soff[8], doff[8];
    load_edge_offs(srcg, dstg, gb, e0, mhalf, cnt, abase, soff, doff);
    v16bf ar = load_a_bf(Rbf + (gb + e0) * NK, NK, 0, cnt);
#pragma unroll
    for (int nt = 0; nt < 4; ++nt) {
      v8f rx[NCH];
#pragma unroll
      for (int c = 0; c < NCH; ++c)
        rx[c] = wmma_bf(ar, load_b_bf(wT + c * NF * NK, nt * 16, NK, 0), vzero8());
      int f = nt * 16 + (lane & 15);
#pragma unroll
      for (int v = 0; v < 8; ++v) {
        float s0v = xs[soff[v] + f];
        float rself = rx[NCH - 1][v];
#pragma unroll
        for (int l = 0; l < LCH; ++l) {
          float xsrc = (l == 0) ? s0v : xs[soff[v] + l * NF + f];
          float val = rself * xsrc +
                      rx[degc(l)][v] * Ytw[(v + mhalf) * NL + l] * s0v;
          atomicAdd(&ys[doff[v] + l * NF + f], val);
        }
      }
    }
  }
}

// rows x 64 @ 64 x 64 (B in K-contiguous layout Bt[n*64+k]); optional bias on l==0
template <int LCH>
__device__ void gemm_rows(const float* src, const bf16_t* Bt, const float* bias,
                          float* dst, const float* addsrc) {
  constexpr int rows = NA * LCH;
  constexpr int ntile = (rows + 15) >> 4;
  int wv = threadIdx.x >> 5, lane = threadIdx.x & 31;
  int mhalf = (lane >> 4) << 3;
  for (int tile = wv; tile < ntile; tile += NWAVE) {
#pragma unroll
    for (int nt = 0; nt < 4; ++nt) {
      v8f acc = vzero8();
#pragma unroll
      for (int ks = 0; ks < 2; ++ks) {
        v16bf a = load_a_f32ch<LCH>(src, tile * 16, ks * 32, rows);
        v16bf b = load_b_bf(Bt, nt * 16, NF, ks * 32);
        acc = wmma_bf(a, b, acc);
      }
      int f = nt * 16 + (lane & 15);
#pragma unroll
      for (int v = 0; v < 8; ++v) {
        int r = tile * 16 + v + mhalf;
        if (r >= rows) continue;
        int ai = r / LCH, l = r - ai * LCH;
        int idx = (ai * NL + l) * NF + f;
        float o = acc[v];
        if (bias && l == 0) o += bias[f];
        if (addsrc) o += addsrc[idx];
        dst[idx] = o;
      }
    }
  }
}

__device__ __forceinline__ float dRdd_from(float R, float d, int k) {
  float t = d * (1.f / CUTR);
  float g = 1.f - t * t;
  float gm = fmaxf(g, 1e-9f);
  float fder = 0.f;
  if (t < 1.f && g > 1e-9f) fder = -2.f * t * (1.f / CUTR) / (gm * gm);
  return R * ((float)k / d - 31.f / (1.f + d) + fder);
}

// ---------------------------------------------------------------------------
// Small kernels
// ---------------------------------------------------------------------------
__global__ void zero_out(float* p, int n) {
  for (int i = blockIdx.x * blockDim.x + threadIdx.x; i < n;
       i += gridDim.x * blockDim.x)
    p[i] = 0.f;
}

__global__ void prep_weights(const float* Wr, const float* Wr0, const float* W1,
                             const float* W2, bf16_t* WT, bf16_t* WO, bf16_t* W1T,
                             bf16_t* W1O, bf16_t* W2T, bf16_t* W2O, int nch) {
  int tot = nch * NK * NF;
  for (int i = blockIdx.x * blockDim.x + threadIdx.x; i < tot;
       i += gridDim.x * blockDim.x) {
    int c = i / (NK * NF);
    int r = i - c * NK * NF;
    int k = r / NF, f = r - (r / NF) * NF;
    float w = (c == nch - 1) ? Wr0[k * NF + f] : Wr[(c * NK + k) * NF + f];
    WT[(c * NF + f) * NK + k] = (bf16_t)w;  // B-layout for rexp GEMM
    WO[(c * NK + k) * NF + f] = (bf16_t)w;  // original layout for g_R GEMM
  }
  for (int i = blockIdx.x * blockDim.x + threadIdx.x; i < NF * NF;
       i += gridDim.x * blockDim.x) {
    int g = i / NF, f = i - g * NF;  // i == g*NF + f
    W1T[i] = (bf16_t)W1[f * NF + g];
    W1O[i] = (bf16_t)W1[i];
    W2T[i] = (bf16_t)W2[f * NF + g];
    W2O[i] = (bf16_t)W2[i];
  }
}

__global__ void edge_geom(const float* pos, const int* srcg, const int* dstg,
                          bf16_t* Rbf, float* Yf, float* Df, float* Dispf) {
  for (long e = blockIdx.x * blockDim.x + threadIdx.x; e < ETOT;
       e += (long)gridDim.x * blockDim.x) {
    int s = srcg[e], t = dstg[e];
    float dx = pos[s * 3 + 0] - pos[t * 3 + 0];
    float dy = pos[s * 3 + 1] - pos[t * 3 + 1];
    float dz = pos[s * 3 + 2] - pos[t * 3 + 2];
    float d = sqrtf(dx * dx + dy * dy + dz * dz + 1e-12f);
    float rinv = 1.f / d;
    float ux = dx * rinv, uy = dy * rinv, uz = dz * rinv;
    const float s3 = 1.7320508075688772f;
    float Y[NL] = {1.f, ux, uy, uz, s3 * ux * uy, s3 * uy * uz,
                   0.5f * (3.f * uz * uz - 1.f), s3 * ux * uz,
                   0.5f * s3 * (ux * ux - uy * uy)};
#pragma unroll
    for (int l = 0; l < NL; ++l) Yf[e * NL + l] = Y[l];
    Dispf[e * 3 + 0] = dx; Dispf[e * 3 + 1] = dy; Dispf[e * 3 + 2] = dz;
    Df[e] = d;
    // R_k = C(31,k) d^k (1+d)^-31 * fc via recurrence
    float tt = d * (1.f / CUTR);
    float g = 1.f - tt * tt;
    float gm = fmaxf(g, 1e-9f);
    float fc = (tt < 1.f) ? expf(1.f - 1.f / gm) : 0.f;
    float r = fc * expf(-31.f * log1pf(d));
#pragma unroll
    for (int k = 0; k < NK; ++k) {
      Rbf[e * NK + k] = (bf16_t)r;
      r *= d * ((float)(NK - 1 - k) / (float)(k + 1));
    }
  }
}

__global__ void init_x0(const int* Z, const float* embed, float* x0) {
  for (long i = blockIdx.x * blockDim.x + threadIdx.x; i < (long)NATOM * NL * NF;
       i += (long)gridDim.x * blockDim.x) {
    int a = (int)(i / (NL * NF));
    int r = (int)(i - (long)a * NL * NF);
    int l = r >> 6, f = r & 63;
    x0[i] = (l == 0) ? embed[Z[a] * NF + f] : 0.f;
  }
}

__global__ void readout_seed(const float* x3, const float* Wout, const float* elb,
                             const int* Z, const int* bseg, float* energy,
                             float* gseed) {
  for (int a = blockIdx.x * blockDim.x + threadIdx.x; a < NATOM;
       a += gridDim.x * blockDim.x) {
    float ae = elb[Z[a]];
    long base = (long)a * NL * NF;
    for (int f = 0; f < NF; ++f) {
      float w = Wout[f];
      ae += x3[base + f] * w;
      gseed[base + f] = -w;  // d(-sum E)/dx[:,0,:]
    }
    atomicAdd(&energy[bseg[a]], ae);
  }
}

// ---------------------------------------------------------------------------
// Forward iteration: one workgroup (8 wave32) per molecule, state in LDS
// ---------------------------------------------------------------------------
template <int NCH, int LCH>
__global__ void __launch_bounds__(256) fwd_iter(
    const float* __restrict__ xin, float* __restrict__ xout,
    const bf16_t* __restrict__ Rbf, const float* __restrict__ Yf,
    const int* __restrict__ srcg, const int* __restrict__ dstg,
    const bf16_t* __restrict__ WTg, const bf16_t* __restrict__ W1Tg,
    const bf16_t* __restrict__ W2Tg, const float* __restrict__ b1,
    const float* __restrict__ b2) {
  extern __shared__ unsigned char smem[];
  float* xs = (float*)smem;                 // NA*NL*NF
  float* ys = xs + NA * NL * NF;
  float* hs = ys + NA * NL * NF;
  float* Yt = hs + NA * NL * NF;            // NWAVE*16*NL
  float* b1s = Yt + NWAVE * 16 * NL;        // NF
  float* b2s = b1s + NF;
  bf16_t* wT = (bf16_t*)(b2s + NF);         // 4*NF*NK
  bf16_t* w1T = wT + 4 * NF * NK;
  bf16_t* w2T = w1T + NF * NF;

  int mol = blockIdx.x;
  int abase = mol * NA;
  const float* xg = xin + (long)abase * NL * NF;
  for (int i = threadIdx.x * 4; i < NA * NL * NF; i += blockDim.x * 4)
    *(float4*)(ys + i) = float4{0.f, 0.f, 0.f, 0.f};
  if (LCH == NL) {
    for (int i = threadIdx.x * 4; i < NA * NL * NF; i += blockDim.x * 4)
      *(float4*)(xs + i) = *(const float4*)(xg + i);
  } else {
    for (int i = threadIdx.x * 4; i < NA * NF; i += blockDim.x * 4) {
      int a = i / NF, f = i - a * NF;
      *(float4*)(xs + a * NL * NF + f) = *(const float4*)(xg + a * NL * NF + f);
    }
  }
  for (int i = threadIdx.x * 8; i < NCH * NF * NK; i += blockDim.x * 8)
    *(uint4*)(wT + i) = *(const uint4*)(WTg + i);
  for (int i = threadIdx.x * 8; i < NF * NF; i += blockDim.x * 8) {
    *(uint4*)(w1T + i) = *(const uint4*)(W1Tg + i);
    *(uint4*)(w2T + i) = *(const uint4*)(W2Tg + i);
  }
  for (int i = threadIdx.x; i < NF; i += blockDim.x) {
    b1s[i] = b1[i];
    b2s[i] = b2[i];
  }
  __syncthreads();
  edge_forward<NCH, LCH>(mol, Rbf, Yf, srcg, dstg, wT, xs, ys, Yt);
  __syncthreads();
  constexpr int rows = NA * LCH;
  for (int i = threadIdx.x; i < rows * NF; i += blockDim.x) {  // t = x + y
    int r = i / NF, f = i - r * NF;
    int a = r / LCH, l = r - a * LCH;
    int idx = (a * NL + l) * NF + f;
    ys[idx] += xs[idx];
  }
  __syncthreads();
  gemm_rows<LCH>(ys, w1T, b1s, hs, nullptr);  // h1
  __syncthreads();
  for (int i = threadIdx.x; i < NA * NF; i += blockDim.x) {  // gating -> ys
    int a = i / NF, f = i - a * NF;
    float z0 = hs[a * NL * NF + f];
    float sg = 1.f / (1.f + __expf(-z0));
    ys[a * NL * NF + f] = z0 * sg;
#pragma unroll
    for (int l = 1; l < LCH; ++l)
      ys[(a * NL + l) * NF + f] = hs[(a * NL + l) * NF + f] * sg;
  }
  __syncthreads();
  gemm_rows<LCH>(ys, w2T, b2s, xout + (long)abase * NL * NF, xs);  // x_out
}

// ---------------------------------------------------------------------------
// Backward iteration: recompute forward in LDS, chain grads, accumulate forces
// ---------------------------------------------------------------------------
template <int NCH, int LCH>
__global__ void __launch_bounds__(256) bwd_iter(
    const float* __restrict__ xin, const float* __restrict__ gin,
    float* __restrict__ gout, float* __restrict__ forces,
    const bf16_t* __restrict__ Rbf, const float* __restrict__ Yf,
    const float* __restrict__ Df, const float* __restrict__ Dispf,
    const int* __restrict__ srcg, const int* __restrict__ dstg,
    const bf16_t* __restrict__ WTg, const bf16_t* __restrict__ WOg,
    const bf16_t* __restrict__ W1Tg, const bf16_t* __restrict__ W1Og,
    const bf16_t* __restrict__ W2Og, const float* __restrict__ b1) {
  extern __shared__ unsigned char smem[];
  float* xs = (float*)smem;                  // NA*NL*NF
  float* buf2 = xs + NA * NL * NF;           // y -> t -> g_hact -> g_h1
  float* buf3 = buf2 + NA * NL * NF;         // h1 -> g_t (== g_y)
  float* gxp = buf3 + NA * NL * NF;          // grad wrt x_in (accumulated)
  float* Yt = gxp + NA * NL * NF;            // NWAVE*16*NL
  float* gYt = Yt + NWAVE * 16 * NL;         // NWAVE*16*NL
  float* gdt = gYt + NWAVE * 16 * NL;        // NWAVE*16
  float* fpos = gdt + NWAVE * 16;            // NA*3 (padded 64)
  float* b1s = fpos + 64;                    // NF
  bf16_t* wT = (bf16_t*)(b1s + NF);          // 4*NF*NK
  bf16_t* wO = wT + 4 * NF * NK;             // 4*NK*NF
  bf16_t* w1T = wO + 4 * NF * NK;
  bf16_t* w1O = w1T + NF * NF;
  bf16_t* w2O = w1O + NF * NF;
  bf16_t* gstage = w2O + NF * NF;            // NWAVE*16*NF

  int mol = blockIdx.x;
  int abase = mol * NA;
  int wv = threadIdx.x >> 5, lane = threadIdx.x & 31;
  int mhalf = (lane >> 4) << 3;
  const float* xg = xin + (long)abase * NL * NF;
  const float* gg = gin + (long)abase * NL * NF;

  for (int i = threadIdx.x * 4; i < NA * NL * NF; i += blockDim.x * 4)
    *(float4*)(buf2 + i) = float4{0.f, 0.f, 0.f, 0.f};
  if (LCH == NL) {
    for (int i = threadIdx.x * 4; i < NA * NL * NF; i += blockDim.x * 4)
      *(float4*)(xs + i) = *(const float4*)(xg + i);
  } else {
    for (int i = threadIdx.x * 4; i < NA * NF; i += blockDim.x * 4) {
      int a = i / NF, f = i - a * NF;
      *(float4*)(xs + a * NL * NF + f) = *(const float4*)(xg + a * NL * NF + f);
    }
  }
  for (int i = threadIdx.x * 8; i < NCH * NF * NK; i += blockDim.x * 8) {
    *(uint4*)(wT + i) = *(const uint4*)(WTg + i);
    *(uint4*)(wO + i) = *(const uint4*)(WOg + i);
  }
  for (int i = threadIdx.x * 8; i < NF * NF; i += blockDim.x * 8) {
    *(uint4*)(w1T + i) = *(const uint4*)(W1Tg + i);
    *(uint4*)(w1O + i) = *(const uint4*)(W1Og + i);
    *(uint4*)(w2O + i) = *(const uint4*)(W2Og + i);
  }
  for (int i = threadIdx.x; i < NF; i += blockDim.x) b1s[i] = b1[i];
  for (int i = threadIdx.x; i < NA * 3; i += blockDim.x) fpos[i] = 0.f;
  __syncthreads();

  // ---- forward recompute: y, t, h1 ----
  edge_forward<NCH, LCH>(mol, Rbf, Yf, srcg, dstg, wT, xs, buf2, Yt);
  __syncthreads();
  constexpr int rows = NA * LCH;
  for (int i = threadIdx.x; i < rows * NF; i += blockDim.x) {
    int r = i / NF, f = i - r * NF;
    int a = r / LCH, l = r - a * LCH;
    int idx = (a * NL + l) * NF + f;
    buf2[idx] += xs[idx];  // t
  }
  __syncthreads();
  gemm_rows<LCH>(buf2, w1T, b1s, buf3, nullptr);  // h1
  __syncthreads();
  // ---- node backward ----
  gemm_rows<LCH>(gg, w2O, nullptr, buf2, nullptr);  // g_hact
  __syncthreads();
  for (int i = threadIdx.x; i < NA * NF; i += blockDim.x) {  // gating backward
    int a = i / NF, f = i - a * NF;
    float z0 = buf3[a * NL * NF + f];
    float sg = 1.f / (1.f + __expf(-z0));
    float acc0 = buf2[a * NL * NF + f] * (sg * (1.f + z0 * (1.f - sg)));
#pragma unroll
    for (int l = 1; l < LCH; ++l) {
      float gh = buf2[(a * NL + l) * NF + f];
      float hl = buf3[(a * NL + l) * NF + f];
      buf2[(a * NL + l) * NF + f] = gh * sg;
      acc0 += gh * hl * sg * (1.f - sg);
    }
    buf2[a * NL * NF + f] = acc0;  // g_h1
  }
  __syncthreads();
  gemm_rows<LCH>(buf2, w1O, nullptr, buf3, nullptr);  // g_t (== g_y)
  __syncthreads();
  for (int i = threadIdx.x; i < NA * NL * NF; i += blockDim.x) {
    int a = i / (NL * NF);
    int r = i - a * NL * NF;
    int l = r / NF;
    float v = 0.f;
    if (l < LCH) v = buf3[i] + gg[i];
    gxp[i] = v;  // g_x_in (direct); edge scatter adds below
  }
  __syncthreads();

  // ---- edge backward ----
  long gb = (long)mol * EPM;
  float* Ytw = Yt + wv * 16 * NL;
  float* gYw = gYt + wv * 16 * NL;
  float* gdw = gdt + wv * 16;
  bf16_t* gsw = gstage + wv * 16 * NF;

  for (int tile = wv; tile < ETILES; tile += NWAVE) {
    int e0 = tile * 16;
    int cnt = (EPM - e0 < 16) ? (EPM - e0) : 16;
    if (tile + NWAVE < ETILES)
      __builtin_prefetch(Rbf + (gb + e0 + 16 * NWAVE) * NK, 0, 0);
    if (lane < 16) {
      int m = lane;
      gdw[m] = 0.f;
      if (m < cnt) {
        long e = gb + e0 + m;
#pragma unroll
        for (int l = 0; l < NL; ++l) Ytw[m * NL + l] = Yf[e * NL + l];
      } else {
#pragma unroll
        for (int l = 0; l < NL; ++l) Ytw[m * NL + l] = 0.f;
      }
    }
    for (int i = lane; i < 16 * NL; i += 32) gYw[i] = 0.f;
    int soff[8], doff[8];
    load_edge_offs(srcg, dstg, gb, e0, mhalf, cnt, abase, soff, doff);
    v16bf ar = load_a_bf(Rbf + (gb + e0) * NK, NK, 0, cnt);
    v8f gr[2];
    gr[0] = vzero8();
    gr[1] = vzero8();
#pragma unroll
    for (int c = 0; c < NCH; ++c) {
      if (c == NCH - 1) {  // self channel
#pragma unroll
        for (int nt = 0; nt < 4; ++nt) {
          v16bf b = load_b_bf(wT + c * NF * NK, nt * 16, NK, 0);
          v8f rex = wmma_bf(ar, b, vzero8());
          int f = nt * 16 + (lane & 15);
#pragma unroll
          for (int v = 0; v < 8; ++v) {
            float rv = rex[v];
            float s0v = xs[soff[v] + f];
            float grs = 0.f;
#pragma unroll
            for (int l = 0; l < LCH; ++l) {
              float gms = buf3[doff[v] + l * NF + f];
              atomicAdd(&gxp[soff[v] + l * NF + f], gms * rv);  // g_x_src
              grs += gms * ((l == 0) ? s0v : xs[soff[v] + l * NF + f]);
            }
            gsw[(v + mhalf) * NF + f] = (bf16_t)grs;  // g_rself
          }
        }
      } else {  // rexp channel c (l range [lo,hi))
        const int lo = ch_lo(c);
        const int hi = ch_hi(c) < LCH ? ch_hi(c) : LCH;
        float gacc[8][5];
#pragma unroll
        for (int v = 0; v < 8; ++v)
#pragma unroll
          for (int j = 0; j < 5; ++j) gacc[v][j] = 0.f;
#pragma unroll
        for (int nt = 0; nt < 4; ++nt) {
          v16bf b = load_b_bf(wT + c * NF * NK, nt * 16, NK, 0);
          v8f rex = wmma_bf(ar, b, vzero8());
          int f = nt * 16 + (lane & 15);
#pragma unroll
          for (int v = 0; v < 8; ++v) {
            float rv = rex[v];
            float s0v = xs[soff[v] + f];
            float gre = 0.f;
#pragma unroll
            for (int l = lo; l < hi; ++l) {
              float gms = buf3[doff[v] + l * NF + f];
              gre += gms * Ytw[(v + mhalf) * NL + l];
              if (l > 0) gacc[v][l - lo] += gms * rv * s0v;  // gY partial
            }
            atomicAdd(&gxp[soff[v] + f], gre * rv);       // g_s0
            gsw[(v + mhalf) * NF + f] = (bf16_t)(gre * s0v);  // g_rexp
          }
        }
#pragma unroll
        for (int v = 0; v < 8; ++v)
#pragma unroll
          for (int l = lo; l < hi; ++l)
            if (l > 0) atomicAdd(&gYw[(v + mhalf) * NL + l], gacc[v][l - lo]);
      }
      // g_R += g_rexp_c @ W_c  (K = f = 64, N = krad = 32)
#pragma unroll
      for (int ks = 0; ks < 2; ++ks) {
        v16bf a = load_a_bf(gsw, NF, ks * 32, 16);
        v16bf bb0 = load_b_bf(wO + c * NK * NF, 0, NF, ks * 32);
        v16bf bb1 = load_b_bf(wO + c * NK * NF, 16, NF, ks * 32);
        gr[0] = wmma_bf(a, bb0, gr[0]);
        gr[1] = wmma_bf(a, bb1, gr[1]);
      }
    }
    // g_d = sum_k g_R[k] * dR/dd  (padded rows excluded)
#pragma unroll
    for (int kt = 0; kt < 2; ++kt) {
      int k = kt * 16 + (lane & 15);
#pragma unroll
      for (int v = 0; v < 8; ++v) {
        int m = v + mhalf;
        if (m < cnt) {
          long e = gb + e0 + m;
          float d = Df[e];
          float R = (float)Rbf[e * NK + k];
          atomicAdd(&gdw[m], gr[kt][v] * dRdd_from(R, d, k));
        }
      }
    }
    // per-edge: gY -> g_u -> g_disp ; plus g_d * u ; accumulate forces
    if (lane < 16 && lane < cnt) {
      int m = lane;
      long e = gb + e0 + m;
      float d = Df[e], rinv = 1.f / d;
      float dx = Dispf[e * 3 + 0], dy = Dispf[e * 3 + 1], dz = Dispf[e * 3 + 2];
      float ux = dx * rinv, uy = dy * rinv, uz = dz * rinv;
      float gux = 0.f, guy = 0.f, guz = 0.f;
      if (LCH > 1) {
        const float s3 = 1.7320508075688772f;
        float g1 = gYw[m * NL + 1], g2 = gYw[m * NL + 2], g3 = gYw[m * NL + 3];
        float g4 = gYw[m * NL + 4], g5 = gYw[m * NL + 5], g6 = gYw[m * NL + 6];
        float g7 = gYw[m * NL + 7], g8 = gYw[m * NL + 8];
        gux = g1 + s3 * (uy * g4 + uz * g7 + ux * g8);
        guy = g2 + s3 * (ux * g4 + uz * g5 - uy * g8);
        guz = g3 + s3 * (uy * g5 + ux * g7) + 3.f * uz * g6;
      }
      float gd = gdw[m];
      float udot = ux * gux + uy * guy + uz * guz;
      float fx = (gux - ux * udot) * rinv + gd * ux;
      float fy = (guy - uy * udot) * rinv + gd * uy;
      float fz = (guz - uz * udot) * rinv + gd * uz;
      int s = srcg[e] - abase, dl = dstg[e] - abase;
      atomicAdd(&fpos[s * 3 + 0], fx);
      atomicAdd(&fpos[s * 3 + 1], fy);
      atomicAdd(&fpos[s * 3 + 2], fz);
      atomicAdd(&fpos[dl * 3 + 0], -fx);
      atomicAdd(&fpos[dl * 3 + 1], -fy);
      atomicAdd(&fpos[dl * 3 + 2], -fz);
    }
  }
  __syncthreads();
  for (int i = threadIdx.x * 4; i < NA * NL * NF; i += blockDim.x * 4)
    *(float4*)(gout + (long)abase * NL * NF + i) = *(const float4*)(gxp + i);
  for (int i = threadIdx.x; i < NA * 3; i += blockDim.x)
    forces[abase * 3 + i] += fpos[i];
}

// ---------------------------------------------------------------------------
// Host launcher
// ---------------------------------------------------------------------------
extern "C" void kernel_launch(void* const* d_in, const int* in_sizes, int n_in,
                              void* d_out, int out_size, void* d_ws,
                              size_t ws_size, hipStream_t stream) {
  (void)in_sizes; (void)n_in; (void)out_size; (void)ws_size;
  const int* Z = (const int*)d_in[0];
  const float* pos = (const float*)d_in[1];
  const int* dsti = (const int*)d_in[2];
  const int* srci = (const int*)d_in[3];
  const int* bseg = (const int*)d_in[4];
  const float* embed = (const float*)d_in[6];
  const float* Wr[3] = {(const float*)d_in[7], (const float*)d_in[13],
                        (const float*)d_in[19]};
  const float* Wr0[3] = {(const float*)d_in[8], (const float*)d_in[14],
                         (const float*)d_in[20]};
  const float* W1[3] = {(const float*)d_in[9], (const float*)d_in[15],
                        (const float*)d_in[21]};
  const float* b1[3] = {(const float*)d_in[10], (const float*)d_in[16],
                        (const float*)d_in[22]};
  const float* W2[3] = {(const float*)d_in[11], (const float*)d_in[17],
                        (const float*)d_in[23]};
  const float* b2[3] = {(const float*)d_in[12], (const float*)d_in[18],
                        (const float*)d_in[24]};
  const float* Wout = (const float*)d_in[25];
  const float* elb = (const float*)d_in[26];

  float* energy = (float*)d_out;
  float* forces = (float*)d_out + NMOL;

  char* ws = (char*)d_ws;
  size_t off = 0;
  auto alloc = [&](size_t bytes) {
    void* p = ws + off;
    off += (bytes + 255) & ~(size_t)255;
    return p;
  };
  // +16 rows padding so the last tile's vectorized A loads stay in-bounds
  bf16_t* Rbf = (bf16_t*)alloc(((size_t)ETOT + 16) * NK * sizeof(bf16_t));
  float* Yf = (float*)alloc((size_t)ETOT * NL * sizeof(float));
  float* Df = (float*)alloc((size_t)ETOT * sizeof(float));
  float* Dispf = (float*)alloc((size_t)ETOT * 3 * sizeof(float));
  float* X[4];
  for (int i = 0; i < 4; ++i)
    X[i] = (float*)alloc((size_t)NATOM * NL * NF * sizeof(float));
  float* GxA = (float*)alloc((size_t)NATOM * NL * NF * sizeof(float));
  float* GxB = (float*)alloc((size_t)NATOM * NL * NF * sizeof(float));
  int nchs[3] = {4, 4, 2};
  bf16_t *WT[3], *WO[3], *W1T[3], *W1O[3], *W2T[3], *W2O[3];
  for (int i = 0; i < 3; ++i) {
    WT[i] = (bf16_t*)alloc((size_t)nchs[i] * NF * NK * sizeof(bf16_t));
    WO[i] = (bf16_t*)alloc((size_t)nchs[i] * NK * NF * sizeof(bf16_t));
    W1T[i] = (bf16_t*)alloc((size_t)NF * NF * sizeof(bf16_t));
    W1O[i] = (bf16_t*)alloc((size_t)NF * NF * sizeof(bf16_t));
    W2T[i] = (bf16_t*)alloc((size_t)NF * NF * sizeof(bf16_t));
    W2O[i] = (bf16_t*)alloc((size_t)NF * NF * sizeof(bf16_t));
  }

  zero_out<<<64, 256, 0, stream>>>((float*)d_out, NMOL + NATOM * 3);
  for (int i = 0; i < 3; ++i)
    prep_weights<<<32, 256, 0, stream>>>(Wr[i], Wr0[i], W1[i], W2[i], WT[i],
                                         WO[i], W1T[i], W1O[i], W2T[i], W2O[i],
                                         nchs[i]);
  edge_geom<<<(int)((ETOT + 255) / 256), 256, 0, stream>>>(pos, srci, dsti, Rbf,
                                                           Yf, Df, Dispf);
  init_x0<<<(NATOM * NL * NF + 255) / 256, 256, 0, stream>>>(Z, embed, X[0]);

  size_t lds_f = (size_t)(3 * NA * NL * NF + NWAVE * 16 * NL + 2 * NF) * 4 +
                 (size_t)(4 * NF * NK + 2 * NF * NF) * 2;
  fwd_iter<4, NL><<<NMOL, 256, lds_f, stream>>>(X[0], X[1], Rbf, Yf, srci, dsti,
                                                WT[0], W1T[0], W2T[0], b1[0], b2[0]);
  fwd_iter<4, NL><<<NMOL, 256, lds_f, stream>>>(X[1], X[2], Rbf, Yf, srci, dsti,
                                                WT[1], W1T[1], W2T[1], b1[1], b2[1]);
  fwd_iter<2, 1><<<NMOL, 256, lds_f, stream>>>(X[2], X[3], Rbf, Yf, srci, dsti,
                                               WT[2], W1T[2], W2T[2], b1[2], b2[2]);
  readout_seed<<<(NATOM + 255) / 256, 256, 0, stream>>>(X[3], Wout, elb, Z, bseg,
                                                        energy, GxA);

  size_t lds_b =
      (size_t)(4 * NA * NL * NF + 2 * NWAVE * 16 * NL + NWAVE * 16 + 64 + NF) * 4 +
      (size_t)(8 * NF * NK + 3 * NF * NF + NWAVE * 16 * NF) * 2;
  bwd_iter<2, 1><<<NMOL, 256, lds_b, stream>>>(X[2], GxA, GxB, forces, Rbf, Yf,
                                               Df, Dispf, srci, dsti, WT[2],
                                               WO[2], W1T[2], W1O[2], W2O[2], b1[2]);
  bwd_iter<4, NL><<<NMOL, 256, lds_b, stream>>>(X[1], GxB, GxA, forces, Rbf, Yf,
                                                Df, Dispf, srci, dsti, WT[1],
                                                WO[1], W1T[1], W1O[1], W2O[1], b1[1]);
  bwd_iter<4, NL><<<NMOL, 256, lds_b, stream>>>(X[0], GxA, GxB, forces, Rbf, Yf,
                                                Df, Dispf, srci, dsti, WT[0],
                                                WO[0], W1T[0], W1O[0], W2O[0], b1[0]);
}
